// ProjectPoint2Image_61795989455183
// MI455X (gfx1250) — matrix operand order repaired and checked by reading, not compiled, not run on hardware
//
#include <hip/hip_runtime.h>

// ---- CDNA5 (gfx1250) wave32 WMMA types ----
typedef __attribute__((ext_vector_type(16))) _Float16 v16h;
typedef __attribute__((ext_vector_type(8)))  _Float16 v8h;
typedef __attribute__((ext_vector_type(8)))  float    v8f;

#define NPTS   1024          // points per batch
#define NCH    16            // feature channels (== WMMA M)
#define WIMG   128
#define HWPIX  (128*128)     // 16384 pixels
#define WAVES  8             // 256 threads / 32
#define TPW    2             // pixel tiles (16 px) per wave, fused in one K-pass
#define BPB    (HWPIX/16/(WAVES*TPW))   // 64 blocks per batch
#define FPAD   8             // +8 halves (16B) row pad -> conflict-free A loads

__global__ __launch_bounds__(256)
void ProjectPoint2Image_kernel(const float* __restrict__ Km,
                               const float* __restrict__ RT,
                               const float* __restrict__ pts3d,
                               const float* __restrict__ feat,
                               const float* __restrict__ scale,
                               float* __restrict__ out)
{
  // per-point params: {u, v, -log2(e)/(scale*sigma^2), 0}
  __shared__ float4   prm[NPTS];                  // 16 KB
  __shared__ _Float16 featT[NCH][NPTS + FPAD];    // ~32.25 KB, f16, transposed

  const int b   = blockIdx.y;
  const int tid = threadIdx.x;

  // K is 3x3 row-major
  const float k00 = Km[0], k01 = Km[1], k02 = Km[2];
  const float k10 = Km[3], k11 = Km[4], k12 = Km[5];
  const float k20 = Km[6], k21 = Km[7], k22 = Km[8];
  const float sigma = k00 * (1.0f / 32.0f);       // N==1024 path
  const float sig2  = sigma * sigma;

  const float* rt = RT + b * 12;                  // 3x4 row-major
  const float r00 = rt[0], r01 = rt[1], r02 = rt[2],  t0 = rt[3];
  const float r10 = rt[4], r11 = rt[5], r12 = rt[6],  t1 = rt[7];
  const float r20 = rt[8], r21 = rt[9], r22 = rt[10], t2 = rt[11];

  // ---------- preprocessing: project points, transpose features ----------
  for (int n = tid; n < NPTS; n += 256) {
    const float* p = pts3d + ((size_t)b * NPTS + n) * 3;
    const float x = p[0], y = p[1], z3 = p[2];
    const float lx = r00 * x + r01 * y + r02 * z3 + t0;
    const float ly = r10 * x + r11 * y + r12 * z3 + t1;
    const float lz = r20 * x + r21 * y + r22 * z3 + t2;
    const float px = k00 * lx + k01 * ly + k02 * lz;
    const float py = k10 * lx + k11 * ly + k12 * lz;
    const float pz = k20 * lx + k21 * ly + k22 * lz;
    const float zc  = fmaxf(pz, 0.1f);
    const float inv = __builtin_amdgcn_rcpf(zc);
    float u = px * inv;
    float v = py * inv;
    if (!(pz > 0.1f)) { u = 1.0e9f; v = 1.0e9f; }   // mask via guaranteed underflow
    const float s    = scale[(size_t)b * NPTS + n];
    const float coef = -1.442695040888963f / (s * sig2);  // fold log2(e) for v_exp_f32
    prm[n] = make_float4(u, v, coef, 0.0f);

    const float* f = feat + ((size_t)b * NPTS + n) * NCH;
#pragma unroll
    for (int c = 0; c < NCH; ++c) featT[c][n] = (_Float16)f[c];
  }
  __syncthreads();

  const int lane = tid & 31;
  const int wave = tid >> 5;
  const int lm   = lane & 15;     // A row (channel) / B column (pixel-in-tile) / D column
  const int hi   = lane >> 4;     // half-wave selector

  // Two 16-pixel tiles per wave, fused into a single pass over all points so
  // the A-fragment and point-param LDS loads are shared between both WMMAs.
  const int tile0 = (blockIdx.x * WAVES + wave) * TPW;       // uniform per wave
  const int pix0  = tile0 * 16 + lm;
  const int pix1  = pix0 + 16;
  const float gv0 = (float)(pix0 >> 7), gu0 = (float)(pix0 & (WIMG - 1));
  const float gv1 = (float)(pix1 >> 7), gu1 = (float)(pix1 & (WIMG - 1));

  v8f acc0 = {}, acc1 = {};

  for (int n0 = 0; n0 < NPTS; n0 += 32) {
    // ---- A fragment (16x32 f16): lane holds channel lm; K = hi*16+{0..15}
    union { v16h v; struct { v8h lo, hi; } h; } A;
    A.h.lo = *(const v8h*)&featT[lm][n0 +     hi * 16];   // ds_load_b128
    A.h.hi = *(const v8h*)&featT[lm][n0 + 8 + hi * 16];   // ds_load_b128

    // ---- cache 16 point params in VGPRs (broadcast ds_load, shared by both tiles)
    const int nb = n0 + hi * 16;                          // element e -> n = nb + e
    float pu[16], pv[16], pc[16];
#pragma unroll
    for (int e = 0; e < 16; ++e) {
      const float4 pr = prm[nb + e];
      pu[e] = pr.x; pv[e] = pr.y; pc[e] = pr.z;
    }

    // ---- B fragments (32x16 f16): lane = pixel column
    v16h B0, B1;
#pragma unroll
    for (int e = 0; e < 16; ++e) {
      const float du0 = pu[e] - gu0, dv0 = pv[e] - gv0;
      const float du1 = pu[e] - gu1, dv1 = pv[e] - gv1;
      const float d0  = du0 * du0 + dv0 * dv0;
      const float d1  = du1 * du1 + dv1 * dv1;
      B0[e] = (_Float16)__builtin_amdgcn_exp2f(pc[e] * d0);  // v_exp_f32
      B1[e] = (_Float16)__builtin_amdgcn_exp2f(pc[e] * d1);
    }

    // D(16c x 16p) += A(16c x 32n) * B(32n x 16p), f32 accumulate; shared A
    acc0 = __builtin_amdgcn_wmma_f32_16x16x32_f16(
        false, A.v, false, B0, (short)0, acc0, false, false);
    acc1 = __builtin_amdgcn_wmma_f32_16x16x32_f16(
        false, A.v, false, B1, (short)0, acc1, false, false);
  }

  // ---- store D tiles: lane(hi,lm), VGPR j -> out[b][hi*8+j][pix]
  float* ob0 = out + (size_t)b * NCH * HWPIX + pix0;
  float* ob1 = out + (size_t)b * NCH * HWPIX + pix1;
#pragma unroll
  for (int j = 0; j < 8; ++j) ob0[(size_t)(hi * 8 + j) * HWPIX] = acc0[j];
#pragma unroll
  for (int j = 0; j < 8; ++j) ob1[(size_t)(hi * 8 + j) * HWPIX] = acc1[j];
}

extern "C" void kernel_launch(void* const* d_in, const int* in_sizes, int n_in,
                              void* d_out, int out_size, void* d_ws, size_t ws_size,
                              hipStream_t stream) {
  const float* Km    = (const float*)d_in[0];   // (3,3)
  const float* RT    = (const float*)d_in[1];   // (B,3,4)
  const float* pts3d = (const float*)d_in[2];   // (B,N,3)
  const float* feat  = (const float*)d_in[3];   // (B,N,16)
  const float* scale = (const float*)d_in[4];   // (B,N)
  const int B = in_sizes[1] / 12;

  dim3 grid(BPB, B);                            // 64 x B blocks, 256 thr (8 wave32)
  ProjectPoint2Image_kernel<<<grid, 256, 0, stream>>>(Km, RT, pts3d, feat, scale,
                                                      (float*)d_out);
}
